// CapsuleNetwork_78400333021448
// MI455X (gfx1250) — matrix operand order repaired and checked by reading, not compiled, not run on hardware
//
#include <hip/hip_runtime.h>
#include <hip/hip_bf16.h>

// ---------------------------------------------------------------------------
// CapsuleNetwork forward for MI455X (gfx1250, wave32, WMMA).
//
// Precision: bf16 WMMA (v_wmma_f32_16x16x32_bf16) with fp32 accumulate for all
// GEMMs. GEMM1 (4096x10000x512 = 42 GFLOP) dominates; roofline: ~185 MB of
// inputs/weights => ~8 us at 23.3 TB/s, bf16 matrix math lands in the same
// ballpark, so GEMM1 is balanced. Everything else is small -- as long as the
// 838 MB preds tensor never touches HBM (fused routing kernel, preds in LDS).
//
// Data movement: double-buffered LDS tiles; A tiles DMA'd raw f32 via
// global_load_async_to_lds_b128 (ASYNCcnt) for tile t+1 while tile t's WMMAs
// run; B tiles staged branch-free (clamped addr + cndmask) as packed bf16.
// One s_wait_asynccnt + one barrier per k-iteration. All 4 B fragments are
// preloaded so the 4 WMMAs issue back-to-back.
// ---------------------------------------------------------------------------

typedef __attribute__((ext_vector_type(16))) __bf16 v16bf;
typedef __attribute__((ext_vector_type(8)))  float  v8f;
typedef int v4i __attribute__((vector_size(16)));               // int4 vector
typedef v4i __attribute__((address_space(1)))* gv4i_p;          // global int4*
typedef v4i __attribute__((address_space(3)))* lv4i_p;          // LDS int4*

union Frag {
    v16bf v;
    unsigned int u[8];
};

// pack two f32 -> two bf16 (RNE): round both, then one v_perm_b32 grabs the
// two high halves ({hi[3],hi[2],lo[3],lo[2]} byte select = 0x07060302).
__device__ __forceinline__ unsigned int f2bfpk(float lo, float hi) {
    unsigned int ul = __float_as_uint(lo), uh = __float_as_uint(hi);
    unsigned int rl = ul + 0x7FFFu + ((ul >> 16) & 1u);
    unsigned int rh = uh + 0x7FFFu + ((uh >> 16) & 1u);
    return __builtin_amdgcn_perm(rh, rl, 0x07060302u);
}

#if __has_builtin(__builtin_amdgcn_global_load_async_to_lds_b128) && \
    __has_builtin(__builtin_amdgcn_s_wait_asynccnt)
#define USE_ASYNC_LDS 1
#else
#define USE_ASYNC_LDS 0
#endif

__device__ __forceinline__ void copy16_g2l(const float* g, float* l) {
#if USE_ASYNC_LDS
    __builtin_amdgcn_global_load_async_to_lds_b128(
        (gv4i_p)(const void*)g, (lv4i_p)(void*)l, 0, 0);
#else
    *(float4*)l = *(const float4*)g;
#endif
}
__device__ __forceinline__ void wait_async_lds() {
#if USE_ASYNC_LDS
    __builtin_amdgcn_s_wait_asynccnt(0);
#endif
}

// ---------------------------------------------------------------------------
// Generic C[M,N] = act(A[M,K] @ B[K,N] + bias), bf16 WMMA, fp32 accum.
// Requires K % 4 == 0 and N % 4 == 0 (true for all layers here).
// Block: 256 threads = 8 waves. Block tile 128(M) x 64(N); each wave owns a
// 16x64 strip = 4 WMMA accumulators. K stepped by 32, double-buffered.
// ---------------------------------------------------------------------------
#define BM 128
#define BN 64
#define KT 32
#define ASTRIDE 36   // f32 per A row in LDS (padded; bank-conflict-free reads)
#define BSTRIDE 18   // uint (bf16 pairs) per B row in LDS (even => 8B aligned)

__device__ __forceinline__ void stage_tile(
    const float* __restrict__ A, const float* __restrict__ B,
    int M, int N, int K, int m0, int n0, int k0,
    float* As, unsigned int* Bs, int tid)
{
    const bool tail = (k0 + KT > K) || (m0 + BM > M);

    // ---- A tile (BM x KT raw f32): async DMA on the fast path ----
    if (!tail) {
        const float* Abase = A + (size_t)m0 * K + k0;
        for (int u = tid; u < BM * (KT / 4); u += 256) {       // 1024 16B chunks
            int row = u >> 3, kg = u & 7;
            copy16_g2l(Abase + (size_t)row * K + kg * 4,
                       &As[row * ASTRIDE + kg * 4]);
        }
    } else {
        for (int u = tid; u < BM * (KT / 4); u += 256) {
            int row = u >> 3, kg = u & 7;
            int gm = m0 + row, gk = k0 + kg * 4;
            bool valid = (gm < M) && (gk < K);                 // K%4==0 => group-valid
            const float* src = A + (size_t)(gm < M ? gm : M - 1) * K
                                 + (gk < K ? gk : 0);
            float4 val = valid ? *(const float4*)src : make_float4(0.f, 0.f, 0.f, 0.f);
            *(float4*)&As[row * ASTRIDE + kg * 4] = val;
        }
    }

    // ---- B tile (KT x BN) -> bf16 pairs [n][k], branch-free ----
    for (int u = tid; u < BN * (KT / 4); u += 256) {           // 512 units
        int col = u & 63, kg = u >> 6;
        int gn = n0 + col;
        int gnc = gn < N ? gn : N - 1;
        float val[4];
        #pragma unroll
        for (int j = 0; j < 4; ++j) {
            int gk  = k0 + kg * 4 + j;
            int gkc = gk < K ? gk : K - 1;
            float x = B[(size_t)gkc * N + gnc];                // always in-bounds
            val[j]  = (gn < N && gk < K) ? x : 0.f;            // v_cndmask, no branch
        }
        uint2 pk = make_uint2(f2bfpk(val[0], val[1]), f2bfpk(val[2], val[3]));
        *(uint2*)&Bs[col * BSTRIDE + kg * 2] = pk;
    }
}

__global__ __launch_bounds__(256)
void gemm_bf16_wmma(const float* __restrict__ A, const float* __restrict__ B,
                    const float* __restrict__ bias, float* __restrict__ C,
                    int M, int N, int K, int relu)
{
    __shared__ float        As_f[2][BM * ASTRIDE];  // raw f32 A tiles (double buf)
    __shared__ unsigned int Bs_u[2][BN * BSTRIDE];  // bf16-pair B tiles (double buf)

    const int tid  = threadIdx.x;
    const int lane = tid & 31;
    const int wave = tid >> 5;
    const int h    = lane >> 4;   // lane half (0/1)
    const int l16  = lane & 15;

    const int m0 = blockIdx.y * BM;
    const int n0 = blockIdx.x * BN;
    const int ktiles = (K + KT - 1) / KT;

    v8f acc[4] = {};

    // prologue: stage tile 0 into buffer 0
    stage_tile(A, B, M, N, K, m0, n0, 0, As_f[0], Bs_u[0], tid);
    wait_async_lds();
    __syncthreads();

    for (int t = 0; t < ktiles; ++t) {
        const float*        As = As_f[t & 1];
        const unsigned int* Bs = Bs_u[t & 1];

        // ---- A fragment: ISA 7.12.2 16-bit A 16x32 layout.
        // VGPR q (g=q>>2,w=q&3), lane half h: K = g*16 + h*8 + w*2 (+0,1)
        // => per (g,h): 8 consecutive f32 => two float4 LDS loads + pack.
        Frag af;
        const float* arow = &As[(wave * 16 + l16) * ASTRIDE];
        #pragma unroll
        for (int g = 0; g < 2; ++g) {
            float4 lo = *(const float4*)&arow[g * 16 + h * 8];
            float4 hi = *(const float4*)&arow[g * 16 + h * 8 + 4];
            af.u[g * 4 + 0] = f2bfpk(lo.x, lo.y);
            af.u[g * 4 + 1] = f2bfpk(lo.z, lo.w);
            af.u[g * 4 + 2] = f2bfpk(hi.x, hi.y);
            af.u[g * 4 + 3] = f2bfpk(hi.z, hi.w);
        }

        // ---- preload all 4 B fragments, then fire WMMAs back-to-back ----
        // B layout matches A: VGPR q, lane half h: K = (q>>2)*16 + h*8 + (q&3)*2 (+0,1).
        Frag bfrag[4];
        #pragma unroll
        for (int tt = 0; tt < 4; ++tt) {
            const unsigned int* brow = &Bs[(tt * 16 + l16) * BSTRIDE];
            #pragma unroll
            for (int q = 0; q < 8; ++q) bfrag[tt].u[q] = brow[(q >> 2) * 8 + h * 4 + (q & 3)];
        }
        #pragma unroll
        for (int tt = 0; tt < 4; ++tt)
            acc[tt] = __builtin_amdgcn_wmma_f32_16x16x32_bf16(
                false, af.v, false, bfrag[tt].v, (short)0, acc[tt], false, false);

        // ---- stage tile t+1 into the other buffer while WMMAs execute ----
        if (t + 1 < ktiles) {
            stage_tile(A, B, M, N, K, m0, n0, (t + 1) * KT,
                       As_f[(t + 1) & 1], Bs_u[(t + 1) & 1], tid);
            if (t + 2 < ktiles)  // speculative prefetch of tile t+2's B rows
                __builtin_prefetch(&B[(size_t)(t + 2) * KT * N + n0 + (tid & 63)], 0, 1);
            wait_async_lds();    // own wave's async copies for t+1 done
            __syncthreads();     // everyone's tile t+1 visible; tile t free to reuse
        }
    }

    // ---- epilogue: bias + optional ReLU; C layout: VGPR r -> M = h*8 + r ----
    #pragma unroll
    for (int t = 0; t < 4; ++t) {
        int n = n0 + t * 16 + l16;
        if (n >= N) continue;
        float bv = bias[n];
        #pragma unroll
        for (int r = 0; r < 8; ++r) {
            int m = m0 + wave * 16 + h * 8 + r;
            if (m < M) {
                float x = acc[t][r] + bv;
                if (relu) x = fmaxf(x, 0.f);
                C[(size_t)m * N + n] = x;
            }
        }
    }
}

// ---------------------------------------------------------------------------
// Repack Wp (n,i,o) -> Wflat[i, n*8+o] so primary caps is a plain GEMM.
// ---------------------------------------------------------------------------
__global__ void repack_wp(const float* __restrict__ Wp, float* __restrict__ out) {
    int idx = blockIdx.x * blockDim.x + threadIdx.x;       // 32*256*8 = 65536
    if (idx >= 32 * 256 * 8) return;
    int n = idx >> 11, i = (idx >> 3) & 255, o = idx & 7;
    out[i * 256 + n * 8 + o] = Wp[idx];                     // Wp[n*2048 + i*8 + o]
}

// ---------------------------------------------------------------------------
// LayerNorm over PRIMARY_DIM=8, in place. One thread per (b, n) capsule.
// ---------------------------------------------------------------------------
__global__ void primary_ln(float* __restrict__ p, const float* __restrict__ g,
                           const float* __restrict__ bb, int total) {
    int idx = blockIdx.x * blockDim.x + threadIdx.x;        // total = 4096*32
    if (idx >= total) return;
    float* row = p + (size_t)idx * 8;
    int n = idx & 31;
    float mu = 0.f;
    #pragma unroll
    for (int o = 0; o < 8; ++o) mu += row[o];
    mu *= 0.125f;
    float var = 0.f;
    #pragma unroll
    for (int o = 0; o < 8; ++o) { float d = row[o] - mu; var += d * d; }
    var *= 0.125f;
    float inv = rsqrtf(var + 1e-5f);
    #pragma unroll
    for (int o = 0; o < 8; ++o)
        row[o] = (row[o] - mu) * inv * g[n * 8 + o] + bb[n * 8 + o];
}

// ---------------------------------------------------------------------------
// Fused preds + dynamic routing. One block (256 threads) per sample.
// LDS: p(256) + preds(51200) + blog(3200) + c(3200) + v(1600) floats = 232 KB
// -> fits CDNA5's 320 KB/WGP LDS; preds never touches HBM.
// ---------------------------------------------------------------------------
__global__ __launch_bounds__(256)
void caps_route(const float* __restrict__ p, const float* __restrict__ Wr,
                float* __restrict__ vout) {
    extern __shared__ float sm[];
    float* p_s   = sm;                 // 256
    float* preds = p_s + 256;          // 32*100*16 = 51200
    float* blog  = preds + 51200;      // 32*100
    float* c_s   = blog + 3200;        // 32*100
    float* v_s   = c_s + 3200;         // 100*16

    const int b = blockIdx.x, tid = threadIdx.x;

    p_s[tid] = p[(size_t)b * 256 + tid];
    for (int i = tid; i < 3200; i += 256) blog[i] = 0.f;
    __syncthreads();

    // preds[n,j,o] = sum_i p[n,i] * Wr[n,j,o,i]   (flat idx = n*1600 + j*16 + o)
    for (int idx = tid; idx < 51200; idx += 256) {
        const float* w  = Wr + (size_t)idx * 8;
        const float* pv = p_s + (idx / 1600) * 8;
        float acc = 0.f;
        #pragma unroll
        for (int i = 0; i < 8; ++i) acc += pv[i] * w[i];
        preds[idx] = acc;
    }

    for (int it = 0; it < 3; ++it) {
        __syncthreads();
        // softmax over j (labels) per primary capsule n
        if (tid < 32) {
            const int n = tid;
            float mx = -1e30f;
            for (int j = 0; j < 100; ++j) mx = fmaxf(mx, blog[n * 100 + j]);
            float s = 0.f;
            for (int j = 0; j < 100; ++j) {
                float e = __expf(blog[n * 100 + j] - mx);
                c_s[n * 100 + j] = e; s += e;
            }
            float inv = 1.f / s;
            for (int j = 0; j < 100; ++j) c_s[n * 100 + j] *= inv;
        }
        __syncthreads();
        // s[j,o] = sum_n preds[n,j,o] * c[n,j]
        for (int idx = tid; idx < 1600; idx += 256) {
            int j = idx >> 4;
            float acc = 0.f;
            #pragma unroll
            for (int n = 0; n < 32; ++n) acc += preds[n * 1600 + idx] * c_s[n * 100 + j];
            v_s[idx] = acc;
        }
        __syncthreads();
        // squash per j
        if (tid < 100) {
            float sq = 0.f;
            #pragma unroll
            for (int o = 0; o < 16; ++o) { float t = v_s[tid * 16 + o]; sq += t * t; }
            float f = sq / (1.f + sq) * rsqrtf(sq + 1e-8f);
            #pragma unroll
            for (int o = 0; o < 16; ++o) v_s[tid * 16 + o] *= f;
        }
        __syncthreads();
        // agreement update
        if (it < 2) {
            for (int idx = tid; idx < 3200; idx += 256) {
                int n = idx / 100, j = idx % 100;
                float acc = 0.f;
                #pragma unroll
                for (int o = 0; o < 16; ++o)
                    acc += preds[n * 1600 + j * 16 + o] * v_s[j * 16 + o];
                blog[idx] += acc;
            }
        }
    }
    __syncthreads();
    for (int idx = tid; idx < 1600; idx += 256)
        vout[(size_t)b * 1600 + idx] = v_s[idx];
}

// ---------------------------------------------------------------------------
extern "C" void kernel_launch(void* const* d_in, const int* in_sizes, int n_in,
                              void* d_out, int out_size, void* d_ws, size_t ws_size,
                              hipStream_t stream) {
    const float* features = (const float*)d_in[0];  // [4096,10000]
    const float* W1  = (const float*)d_in[1];       // [10000,512]
    const float* b1  = (const float*)d_in[2];       // [512]
    const float* W2  = (const float*)d_in[3];       // [512,256]
    const float* b2  = (const float*)d_in[4];       // [256]
    const float* Wp  = (const float*)d_in[5];       // [32,256,8]
    const float* bp  = (const float*)d_in[6];       // [32,8] -> flat 256 bias
    const float* lng = (const float*)d_in[7];       // [32,8]
    const float* lnb = (const float*)d_in[8];       // [32,8]
    const float* Wr  = (const float*)d_in[9];       // [32,100,16,8]
    const float* Wc  = (const float*)d_in[10];      // [1600,100]
    const float* bc  = (const float*)d_in[11];      // [100]
    float* out = (float*)d_out;                     // [4096,100]

    const int B = 4096;
    float* ws    = (float*)d_ws;
    float* x1    = ws;                              // 4096*512
    float* x2    = x1 + (size_t)B * 512;            // 4096*256
    float* pbuf  = x2 + (size_t)B * 256;            // 4096*256
    float* wpf   = pbuf + (size_t)B * 256;          // 256*256
    float* vcaps = wpf + 256 * 256;                 // 4096*1600

    // 1) x1 = relu(features @ W1 + b1)   M=4096 K=10000 N=512
    gemm_bf16_wmma<<<dim3((512 + BN - 1) / BN, (B + BM - 1) / BM), 256, 0, stream>>>(
        features, W1, b1, x1, B, 512, 10000, 1);
    // 2) x2 = relu(x1 @ W2 + b2)         M=4096 K=512 N=256
    gemm_bf16_wmma<<<dim3((256 + BN - 1) / BN, (B + BM - 1) / BM), 256, 0, stream>>>(
        x1, W2, b2, x2, B, 256, 512, 1);
    // 3) repack Wp for plain GEMM, then p = x2 @ Wpflat + bp
    repack_wp<<<(65536 + 255) / 256, 256, 0, stream>>>(Wp, wpf);
    gemm_bf16_wmma<<<dim3((256 + BN - 1) / BN, (B + BM - 1) / BM), 256, 0, stream>>>(
        x2, wpf, bp, pbuf, B, 256, 256, 0);
    // 4) LayerNorm over the 8-wide capsule dim (in place)
    primary_ln<<<(B * 32 + 255) / 256, 256, 0, stream>>>(pbuf, lng, lnb, B * 32);
    // 5) fused preds + 3 routing iterations, preds lives in 232 KB LDS
    const size_t route_smem = (size_t)(256 + 51200 + 3200 + 3200 + 1600) * sizeof(float);
    caps_route<<<B, 256, route_smem, stream>>>(pbuf, Wr, vcaps);
    // 6) logits = vcaps @ Wc + bc        M=4096 K=1600 N=100
    gemm_bf16_wmma<<<dim3((100 + BN - 1) / BN, (B + BM - 1) / BM), 256, 0, stream>>>(
        vcaps, Wc, bc, out, B, 100, 1600, 0);
    (void)in_sizes; (void)n_in; (void)out_size; (void)ws_size;
}
